// DeformableTransformer_55319178772976
// MI455X (gfx1250) — matrix operand order repaired and checked by reading, not compiled, not run on hardware
//
#include <hip/hip_runtime.h>
#include <math.h>

typedef __bf16 bf16_t;
typedef bf16_t v16bf __attribute__((ext_vector_type(16)));
typedef bf16_t v8bf  __attribute__((ext_vector_type(8)));
typedef bf16_t v2bf  __attribute__((ext_vector_type(2)));
typedef float  v8f   __attribute__((ext_vector_type(8)));

#define NB      8
#define LQ      300
#define DM      256
#define DFF     1024
#define NH      8
#define NLVL    4
#define NPT     4
#define LENTOT  21760
#define NROWS   (NB*LQ)          // 2400
#define LDSSTR  40               // padded bf16 stride (80B rows -> 16B-aligned chunks)

__device__ inline v8bf pack8(const float4& f0, const float4& f1)
{
    v8bf o;
    o[0] = (bf16_t)f0.x; o[1] = (bf16_t)f0.y; o[2] = (bf16_t)f0.z; o[3] = (bf16_t)f0.w;
    o[4] = (bf16_t)f1.x; o[5] = (bf16_t)f1.y; o[6] = (bf16_t)f1.z; o[7] = (bf16_t)f1.w;
    return o;
}
__device__ inline v2bf pack2(float a, float b)
{
    v2bf o;
    o[0] = (bf16_t)a; o[1] = (bf16_t)b;     // -> one v_cvt_pk_bf16_f32
    return o;
}

// ---------------------------------------------------------------------------
// Generic batched WMMA GEMM:  C[z] = act(A[z] (K) * B[z] + bias)
// A fp32 [M,K] (lda), B fp32 [K,N] (ldb) or [N,K] if BT, C fp32/bf16 [M,N] (ldc)
// batch z: outer = z/zdiv, inner = z%zdiv; base offsets sXo*outer + sXi*inner
// 64x64 block tile, 8 waves (4x2), two 16x16 WMMA accumulators per wave,
// K-step 32, bf16 LDS tiles, double-buffered.
//
// OOB policy: A rows / B cols are CLAMPED (garbage only reaches C rows/cols
// that the guarded epilogue never stores). Only the K-tail is zero-filled.
// => one branch-free main loop for every block.
// ---------------------------------------------------------------------------
template<bool BT, bool RELU, bool OUT_BF16>
__global__ __launch_bounds__(256)
void gemm_wmma(const float* __restrict__ A, int lda, long long sAo, long long sAi,
               const float* __restrict__ B, int ldb, long long sBo, long long sBi,
               const float* __restrict__ bias,
               void* __restrict__ Cv, int ldc, long long sCo, long long sCi,
               int M, int N, int K, int zdiv)
{
    __shared__ bf16_t As[2][64 * LDSSTR];
    __shared__ bf16_t Bs[2][64 * LDSSTR];

    const int z  = blockIdx.z;
    const int zo = z / zdiv, zi = z % zdiv;
    A += zo * sAo + zi * sAi;
    B += zo * sBo + zi * sBi;
    const long long coff = (long long)zo * sCo + (long long)zi * sCi;

    const int t  = threadIdx.x;
    const int w  = t >> 5, l = t & 31;
    const int wm = w & 3, wn = w >> 2;          // 4x2 wave grid over 64x64 tile
    const int bm = blockIdx.x * 64, bn = blockIdx.y * 64;
    const int hsel = l >> 4, l16 = l & 15;
    const int abk  = hsel * 8;                   // A frag K-chunk base (ISA 16-bit A layout)
    const int bbk  = hsel * 8;                   // B frag K-chunk base (same layout as A)

    // loader thread mappings
    const int ar = t >> 2, aseg = (t & 3) * 8;   // A: row, k-segment
    const int bnr = t >> 2, bks = (t & 3) * 8;   // B (BT): n-row, k-segment
    const int bkp = (t & 15) * 2;                // B (KN): k-pair base
    const int bnp = (t >> 4) * 4;                // B (KN): 4-column group

    // clamped per-thread base pointers (hoisted out of the K loop)
    const int gr  = bm + ar;
    const int grc = gr < M ? gr : M - 1;
    const float* Arow = A + (size_t)grc * lda;
    const float* Brow = nullptr;                 // BT only
    if (BT) {
        const int gn  = bn + bnr;
        const int gnc = gn < N ? gn : N - 1;
        Brow = B + (size_t)gnc * ldb;
    }

    // ---- full-tile loaders (no guards at all) ------------------------------
    auto loadA = [&](int buf, int kb) {
        const float* p = Arow + kb + aseg;
        *(v8bf*)&As[buf][ar * LDSSTR + aseg] =
            pack8(*(const float4*)p, *(const float4*)(p + 4));
    };
    auto loadB = [&](int buf, int kb) {
        if (BT) {
            const float* p = Brow + kb + bks;
            *(v8bf*)&Bs[buf][bnr * LDSSTR + bks] =
                pack8(*(const float4*)p, *(const float4*)(p + 4));
        } else {
            // two consecutive k-rows x 4 columns; write packed (k,k+1) pairs
            const float* p0 = B + (size_t)(kb + bkp) * ldb + bn + bnp;
            const float4 f0 = *(const float4*)p0;
            const float4 f1 = *(const float4*)(p0 + ldb);
            bf16_t* dst = &Bs[buf][bnp * LDSSTR + bkp];
            *(v2bf*)(dst + 0 * LDSSTR) = pack2(f0.x, f1.x);
            *(v2bf*)(dst + 1 * LDSSTR) = pack2(f0.y, f1.y);
            *(v2bf*)(dst + 2 * LDSSTR) = pack2(f0.z, f1.z);
            *(v2bf*)(dst + 3 * LDSSTR) = pack2(f0.w, f1.w);
        }
    };
    // ---- K-tail loaders (zero-fill k >= K; loads batched, then select) -----
    auto loadA_tail = [&](int buf, int kb) {
        float vals[8];
        #pragma unroll
        for (int j = 0; j < 8; ++j) {
            int k = kb + aseg + j;
            vals[j] = Arow[k < K ? k : K - 1];
        }
        v8bf o;
        #pragma unroll
        for (int j = 0; j < 8; ++j)
            o[j] = (bf16_t)((kb + aseg + j) < K ? vals[j] : 0.0f);
        *(v8bf*)&As[buf][ar * LDSSTR + aseg] = o;
    };
    auto loadB_tail = [&](int buf, int kb) {
        if (BT) {
            float vals[8];
            #pragma unroll
            for (int j = 0; j < 8; ++j) {
                int k = kb + bks + j;
                vals[j] = Brow[k < K ? k : K - 1];
            }
            v8bf o;
            #pragma unroll
            for (int j = 0; j < 8; ++j)
                o[j] = (bf16_t)((kb + bks + j) < K ? vals[j] : 0.0f);
            *(v8bf*)&Bs[buf][bnr * LDSSTR + bks] = o;
        } else {
            const int gk0 = kb + bkp, gk1 = gk0 + 1;
            const float* p0 = B + (size_t)(gk0 < K ? gk0 : K - 1) * ldb + bn + bnp;
            const float* p1 = B + (size_t)(gk1 < K ? gk1 : K - 1) * ldb + bn + bnp;
            const float4 f0 = *(const float4*)p0;
            const float4 f1 = *(const float4*)p1;
            const float a0 = gk0 < K ? f0.x : 0.f, a1 = gk0 < K ? f0.y : 0.f;
            const float a2 = gk0 < K ? f0.z : 0.f, a3 = gk0 < K ? f0.w : 0.f;
            const float c0 = gk1 < K ? f1.x : 0.f, c1 = gk1 < K ? f1.y : 0.f;
            const float c2 = gk1 < K ? f1.z : 0.f, c3 = gk1 < K ? f1.w : 0.f;
            bf16_t* dst = &Bs[buf][bnp * LDSSTR + bkp];
            *(v2bf*)(dst + 0 * LDSSTR) = pack2(a0, c0);
            *(v2bf*)(dst + 1 * LDSSTR) = pack2(a1, c1);
            *(v2bf*)(dst + 2 * LDSSTR) = pack2(a2, c2);
            *(v2bf*)(dst + 3 * LDSSTR) = pack2(a3, c3);
        }
    };

    // ---- MMA ---------------------------------------------------------------
    v8f acc0 = {};
    v8f acc1 = {};
    auto compute = [&](int buf) {
        const bf16_t* Ab = As[buf];
        const bf16_t* Bb = Bs[buf];
        union U { v16bf v; v8bf h[2]; };
        U a, b0, b1;
        // A 16x32 bf16 frag: lane row = l16; lanes<16: K 0-7 & 16-23, lanes>=16: +8
        a.h[0]  = *(const v8bf*)(Ab + (wm * 16 + l16) * LDSSTR + abk);
        a.h[1]  = *(const v8bf*)(Ab + (wm * 16 + l16) * LDSSTR + abk + 16);
        // B 32x16 frag: lane col = l16; lanes<16: K 0-7 & 16-23, lanes>=16: +8
        b0.h[0] = *(const v8bf*)(Bb + (wn * 32 +      l16) * LDSSTR + bbk);
        b0.h[1] = *(const v8bf*)(Bb + (wn * 32 +      l16) * LDSSTR + bbk + 16);
        b1.h[0] = *(const v8bf*)(Bb + (wn * 32 + 16 + l16) * LDSSTR + bbk);
        b1.h[1] = *(const v8bf*)(Bb + (wn * 32 + 16 + l16) * LDSSTR + bbk + 16);
        acc0 = __builtin_amdgcn_wmma_f32_16x16x32_bf16(false, a.v, false, b0.v,
                                                       (short)0, acc0, false, false);
        acc1 = __builtin_amdgcn_wmma_f32_16x16x32_bf16(false, a.v, false, b1.v,
                                                       (short)0, acc1, false, false);
    };

    // ---- main loop: identical for every block ------------------------------
    const int nk = K >> 5;                        // full 32-wide tiles
    int buf = 0;
    if (nk > 0) {
        loadA(0, 0);
        loadB(0, 0);
        for (int it = 1; it < nk; ++it) {         // branch-free steady state
            __syncthreads();
            loadA(buf ^ 1, it << 5);
            loadB(buf ^ 1, it << 5);
            compute(buf);
            buf ^= 1;
        }
        __syncthreads();
        compute(buf);
        buf ^= 1;
    }
    if (K & 31) {                                 // K-tail tile (e.g. K=300)
        __syncthreads();
        loadA_tail(buf, K & ~31);
        loadB_tail(buf, K & ~31);
        __syncthreads();
        compute(buf);
    }

    // ---- epilogue: lane col = l16, VGPR r -> row r + 8*hsel ----------------
    const int gn0 = bn + wn * 32 + l16;
    const int gn1 = gn0 + 16;
    float bv0 = (bias && gn0 < N) ? bias[gn0] : 0.0f;
    float bv1 = (bias && gn1 < N) ? bias[gn1] : 0.0f;

    if ((bm + 64 <= M) && (bn + 64 <= N)) {       // interior: no guards
        #pragma unroll
        for (int r = 0; r < 8; ++r) {
            const int gm = bm + wm * 16 + hsel * 8 + r;
            float v0 = acc0[r] + bv0;
            float v1 = acc1[r] + bv1;
            if (RELU) { v0 = fmaxf(v0, 0.0f); v1 = fmaxf(v1, 0.0f); }
            const long long i0 = coff + (long long)gm * ldc + gn0;
            if (OUT_BF16) {
                ((bf16_t*)Cv)[i0]      = (bf16_t)v0;
                ((bf16_t*)Cv)[i0 + 16] = (bf16_t)v1;
            } else {
                ((float*)Cv)[i0]      = v0;
                ((float*)Cv)[i0 + 16] = v1;
            }
        }
    } else {
        #pragma unroll
        for (int r = 0; r < 8; ++r) {
            const int gm = bm + wm * 16 + hsel * 8 + r;
            if (gm >= M) continue;
            #pragma unroll
            for (int tt = 0; tt < 2; ++tt) {
                const int gn = tt ? gn1 : gn0;
                if (gn >= N) continue;
                float v = (tt ? acc1[r] : acc0[r]) + (tt ? bv1 : bv0);
                if (RELU) v = fmaxf(v, 0.0f);
                const long long idx = coff + (long long)gm * ldc + gn;
                if (OUT_BF16) ((bf16_t*)Cv)[idx] = (bf16_t)v;
                else          ((float*)Cv)[idx]  = v;
            }
        }
    }
}

// ---------------------------------------------------------------------------
__device__ inline float wave_sum(float v) {
    #pragma unroll
    for (int o = 16; o > 0; o >>= 1) v += __shfl_xor(v, o, 32);
    return v;
}
__device__ inline float wave_max(float v) {
    #pragma unroll
    for (int o = 16; o > 0; o >>= 1) v = fmaxf(v, __shfl_xor(v, o, 32));
    return v;
}

// out = LayerNorm(res + x) * g + b   (row = block, C = 256 = blockDim)
__global__ __launch_bounds__(256)
void ln_res_kernel(const float* __restrict__ res, const float* __restrict__ x,
                   const float* __restrict__ g, const float* __restrict__ b,
                   float* __restrict__ out)
{
    const int row = blockIdx.x, t = threadIdx.x;
    const size_t base = (size_t)row * DM;
    float v = res[base + t] + x[base + t];
    float s  = wave_sum(v);
    float sq = wave_sum(v * v);
    __shared__ float ps[8], pq[8];
    int w = t >> 5, l = t & 31;
    if (l == 0) { ps[w] = s; pq[w] = sq; }
    __syncthreads();
    float S = 0.f, Q = 0.f;
    #pragma unroll
    for (int i = 0; i < 8; ++i) { S += ps[i]; Q += pq[i]; }
    float mean = S * (1.0f / DM);
    float var  = Q * (1.0f / DM) - mean * mean;
    out[base + t] = (v - mean) * rsqrtf(var + 1e-5f) * g[t] + b[t];
}

// softmax over rows (scale applied to logits); one wave per row
__global__ __launch_bounds__(256)
void softmax_kernel(float* __restrict__ s, int rows, int cols, float scale)
{
    const int w = threadIdx.x >> 5, l = threadIdx.x & 31;
    const int row = blockIdx.x * 8 + w;
    if (row >= rows) return;
    float* p = s + (size_t)row * cols;
    float m = -1e30f;
    for (int i = l; i < cols; i += 32) m = fmaxf(m, p[i] * scale);
    m = wave_max(m);
    float sum = 0.f;
    for (int i = l; i < cols; i += 32) {
        float e = __expf(p[i] * scale - m);
        p[i] = e; sum += e;
    }
    sum = wave_sum(sum);
    float inv = 1.0f / sum;
    for (int i = l; i < cols; i += 32) p[i] *= inv;
}

__global__ void add4_kernel(float4* __restrict__ o, const float4* __restrict__ a,
                            const float4* __restrict__ b, int n4)
{
    int i = blockIdx.x * blockDim.x + threadIdx.x;
    if (i < n4) {
        float4 x = a[i], y = b[i];
        o[i] = make_float4(x.x + y.x, x.y + y.y, x.z + y.z, x.w + y.w);
    }
}

__global__ void copy4_kernel(float4* __restrict__ o, const float4* __restrict__ a, int n4)
{
    int i = blockIdx.x * blockDim.x + threadIdx.x;
    if (i < n4) o[i] = a[i];
}

// ---------------------------------------------------------------------------
// MS deformable-attention sampling.
// block = one (b,q); wave = one head; lane = one channel d of 32.
// value: bf16 (NB, LENTOT, NH, 32); off: (2400, NH*L*P*2); awl: (2400, NH*L*P)
// ---------------------------------------------------------------------------
__global__ __launch_bounds__(256)
void msda_sample_kernel(const bf16_t* __restrict__ value, const float* __restrict__ offb,
                        const float* __restrict__ awl, const float* __restrict__ refp,
                        const float* __restrict__ validr, float* __restrict__ out)
{
    const int bq = blockIdx.x;
    const int h  = threadIdx.x >> 5, d = threadIdx.x & 31;
    const int b  = bq / LQ;
    const float rx = refp[bq * 2 + 0];
    const float ry = refp[bq * 2 + 1];
    const float* op = offb + (size_t)bq * (NH * NLVL * NPT * 2) + h * (NLVL * NPT * 2);
    const float* ap = awl  + (size_t)bq * (NH * NLVL * NPT)     + h * (NLVL * NPT);

    // softmax over the 16 attention-weight logits
    float logit[NLVL * NPT];
    float m = -1e30f;
    #pragma unroll
    for (int i = 0; i < NLVL * NPT; ++i) { logit[i] = ap[i]; m = fmaxf(m, logit[i]); }
    float ssum = 0.f;
    #pragma unroll
    for (int i = 0; i < NLVL * NPT; ++i) { logit[i] = __expf(logit[i] - m); ssum += logit[i]; }
    const float inv = 1.0f / ssum;

    const int HS[NLVL] = {128, 64, 32, 16};
    const int ST[NLVL] = {0, 16384, 20480, 21504};

    float acc = 0.f;
    #pragma unroll
    for (int lv = 0; lv < NLVL; ++lv) {
        const int W = HS[lv], H = HS[lv];
        const float vx = validr[b * (NLVL * 2) + lv * 2 + 0];
        const float vy = validr[b * (NLVL * 2) + lv * 2 + 1];
        #pragma unroll
        for (int p = 0; p < NPT; ++p) {
            const int i = lv * NPT + p;
            const float aw = logit[i] * inv;
            // loc = ref*valid + off/(W,H); x = loc_x*W - 0.5 -> ref*valid*W + off - 0.5
            const float x = (rx * vx) * (float)W + op[i * 2 + 0] - 0.5f;
            const float y = (ry * vy) * (float)H + op[i * 2 + 1] - 0.5f;
            const float x0f = floorf(x), y0f = floorf(y);
            const float wx = x - x0f, wy = y - y0f;
            const int x0 = (int)x0f, y0 = (int)y0f;
            #pragma unroll
            for (int dy = 0; dy < 2; ++dy) {
                #pragma unroll
                for (int dx = 0; dx < 2; ++dx) {
                    const int xi = x0 + dx, yi = y0 + dy;
                    if (xi >= 0 && xi < W && yi >= 0 && yi < H) {
                        const float wgt = (dx ? wx : 1.f - wx) * (dy ? wy : 1.f - wy);
                        const size_t vi =
                            ((size_t)(b * LENTOT + ST[lv] + yi * W + xi) * NH + h) * 32 + d;
                        acc += aw * wgt * (float)value[vi];
                    }
                }
            }
        }
    }
    out[(size_t)bq * DM + h * 32 + d] = acc;
}

// ---------------------------------------------------------------------------
extern "C" void kernel_launch(void* const* d_in, const int* in_sizes, int n_in,
                              void* d_out, int out_size, void* d_ws, size_t ws_size,
                              hipStream_t stream)
{
    (void)in_sizes; (void)n_in; (void)out_size; (void)ws_size;
    const float* tgt_in = (const float*)d_in[0];
    const float* qpos   = (const float*)d_in[1];
    const float* refp   = (const float*)d_in[2];
    const float* src    = (const float*)d_in[3];
    const float* validr = (const float*)d_in[4];
    const float* w_in   = (const float*)d_in[5];
    const float* b_in   = (const float*)d_in[6];
    const float* w_out  = (const float*)d_in[7];
    const float* b_out  = (const float*)d_in[8];
    const float* w_off  = (const float*)d_in[9];
    const float* b_off  = (const float*)d_in[10];
    const float* w_aw   = (const float*)d_in[11];
    const float* b_aw   = (const float*)d_in[12];
    const float* w_val  = (const float*)d_in[13];
    const float* b_val  = (const float*)d_in[14];
    const float* w_op   = (const float*)d_in[15];
    const float* b_op   = (const float*)d_in[16];
    const float* g1     = (const float*)d_in[17];
    const float* bn1    = (const float*)d_in[18];
    const float* g2     = (const float*)d_in[19];
    const float* bn2    = (const float*)d_in[20];
    const float* g3     = (const float*)d_in[21];
    const float* bn3    = (const float*)d_in[22];
    const float* w_f1   = (const float*)d_in[23];
    const float* b_f1   = (const float*)d_in[24];
    const float* w_f2   = (const float*)d_in[25];
    const float* b_f2   = (const float*)d_in[26];
    float* outp = (float*)d_out;

    // workspace carve-up
    size_t o = 0;
    auto give = [&](size_t bytes) -> void* {
        void* p = (char*)d_ws + o;
        o += (bytes + 255) & ~(size_t)255;
        return p;
    };
    const size_t MVAL = (size_t)NB * LENTOT;           // 174080 value rows
    bf16_t* value  = (bf16_t*)give(MVAL * DM * sizeof(bf16_t));
    float*  tgtb   = (float*)give((size_t)NROWS * DM * 4);
    float*  qbuf   = (float*)give((size_t)NROWS * DM * 4);
    float*  qk     = (float*)give((size_t)NROWS * 2 * DM * 4);     // qp|kp, 512 cols
    float*  vv     = (float*)give((size_t)NROWS * DM * 4);
    float*  scores = (float*)give((size_t)NB * NH * LQ * LQ * 4);  // 23 MB
    float*  ctx    = (float*)give((size_t)NROWS * DM * 4);
    float*  tmp    = (float*)give((size_t)NROWS * DM * 4);
    float*  offb   = (float*)give((size_t)NROWS * DM * 4);         // NH*L*P*2 = 256
    float*  awlb   = (float*)give((size_t)NROWS * 128 * 4);
    float*  h1     = (float*)give((size_t)NROWS * DFF * 4);

    const int nel  = NROWS * DM;                      // 614400
    const int nel4 = nel / 4;
    const dim3 blk(256);
    auto gdim = [](int M, int N, int Z) { return dim3((M + 63) / 64, (N + 63) / 64, Z); };

    // working copy of tgt (must not mutate inputs)
    copy4_kernel<<<(nel4 + 255) / 256, blk, 0, stream>>>(
        (float4*)tgtb, (const float4*)tgt_in, nel4);

    // layer-invariant: value = src @ w_val + b_val  -> bf16 (NB, LENTOT, NH, 32)
    gemm_wmma<false, false, true><<<gdim((int)MVAL, DM, 1), blk, 0, stream>>>(
        src, DM, 0, 0, w_val, DM, 0, 0, b_val, value, DM, 0, 0,
        (int)MVAL, DM, DM, 1);

    const long long sQK = (long long)LQ * 512;        // per-batch stride in qk
    const long long sSC = (long long)LQ * LQ;         // per-(b,h) stride in scores
    const long long sVV = (long long)LQ * DM;         // per-batch stride in vv/ctx

    for (int layer = 0; layer < 6; ++layer) {
        // ---- self attention ----
        add4_kernel<<<(nel4 + 255) / 256, blk, 0, stream>>>(
            (float4*)qbuf, (const float4*)tgtb, (const float4*)qpos, nel4);
        // q/k projections: qk = (tgt+pos) @ w_in[:, :512]
        gemm_wmma<false, false, false><<<gdim(NROWS, 512, 1), blk, 0, stream>>>(
            qbuf, DM, 0, 0, w_in, 3 * DM, 0, 0, b_in, qk, 512, 0, 0,
            NROWS, 512, DM, 1);
        // v projection: vv = tgt @ w_in[:, 512:]
        gemm_wmma<false, false, false><<<gdim(NROWS, DM, 1), blk, 0, stream>>>(
            tgtb, DM, 0, 0, w_in + 2 * DM, 3 * DM, 0, 0, b_in + 2 * DM,
            vv, DM, 0, 0, NROWS, DM, DM, 1);
        // scores[b,h] = qp . kp^T   (batched over 64 (b,h))
        gemm_wmma<true, false, false><<<gdim(LQ, LQ, NB * NH), blk, 0, stream>>>(
            qk, 512, sQK, 32, qk + DM, 512, sQK, 32, nullptr,
            scores, LQ, (long long)NH * sSC, sSC, LQ, LQ, 32, NH);
        softmax_kernel<<<(NB * NH * LQ + 7) / 8, blk, 0, stream>>>(
            scores, NB * NH * LQ, LQ, 0.17677669529663687f);  // 1/sqrt(32)
        // ctx[b,q,h,:] = probs . vp
        gemm_wmma<false, false, false><<<gdim(LQ, 32, NB * NH), blk, 0, stream>>>(
            scores, LQ, (long long)NH * sSC, sSC, vv, DM, sVV, 32, nullptr,
            ctx, DM, sVV, 32, LQ, 32, LQ, NH);
        // out projection + residual LN
        gemm_wmma<false, false, false><<<gdim(NROWS, DM, 1), blk, 0, stream>>>(
            ctx, DM, 0, 0, w_out, DM, 0, 0, b_out, tmp, DM, 0, 0,
            NROWS, DM, DM, 1);
        ln_res_kernel<<<NROWS, blk, 0, stream>>>(tgtb, tmp, g2, bn2, tgtb);

        // ---- deformable cross attention ----
        add4_kernel<<<(nel4 + 255) / 256, blk, 0, stream>>>(
            (float4*)qbuf, (const float4*)tgtb, (const float4*)qpos, nel4);
        gemm_wmma<false, false, false><<<gdim(NROWS, DM, 1), blk, 0, stream>>>(
            qbuf, DM, 0, 0, w_off, DM, 0, 0, b_off, offb, DM, 0, 0,
            NROWS, DM, DM, 1);
        gemm_wmma<false, false, false><<<gdim(NROWS, 128, 1), blk, 0, stream>>>(
            qbuf, DM, 0, 0, w_aw, 128, 0, 0, b_aw, awlb, 128, 0, 0,
            NROWS, 128, DM, 1);
        msda_sample_kernel<<<NROWS, blk, 0, stream>>>(value, offb, awlb, refp, validr, ctx);
        gemm_wmma<false, false, false><<<gdim(NROWS, DM, 1), blk, 0, stream>>>(
            ctx, DM, 0, 0, w_op, DM, 0, 0, b_op, tmp, DM, 0, 0,
            NROWS, DM, DM, 1);
        ln_res_kernel<<<NROWS, blk, 0, stream>>>(tgtb, tmp, g1, bn1, tgtb);

        // ---- FFN ----
        gemm_wmma<false, true, false><<<gdim(NROWS, DFF, 1), blk, 0, stream>>>(
            tgtb, DM, 0, 0, w_f1, DFF, 0, 0, b_f1, h1, DFF, 0, 0,
            NROWS, DFF, DM, 1);
        gemm_wmma<false, false, false><<<gdim(NROWS, DM, 1), blk, 0, stream>>>(
            h1, DFF, 0, 0, w_f2, DM, 0, 0, b_f2, tmp, DM, 0, 0,
            NROWS, DM, DFF, 1);
        ln_res_kernel<<<NROWS, blk, 0, stream>>>(tgtb, tmp, g3, bn3, tgtb);
    }

    copy4_kernel<<<(nel4 + 255) / 256, blk, 0, stream>>>(
        (float4*)outp, (const float4*)tgtb, nel4);
}
